// MaskedAutoencoderMamba_53051436040367
// MI455X (gfx1250) — matrix operand order repaired and checked by reading, not compiled, not run on hardware
//
#include <hip/hip_runtime.h>
#include <hip/hip_bf16.h>

// ---------------- problem constants ----------------
static constexpr int Bn    = 4;
static constexpr int GRIDN = 12;
static constexpr int DM    = 384;
static constexpr int DEPTH = 8;
static constexpr int Lt    = GRIDN * GRIDN * GRIDN;   // 1728 voxels
static constexpr int DI    = 768;
static constexpr int NS    = 16;
static constexpr int DC    = 4;
static constexpr int DTR   = 24;
static constexpr int KEEP  = 432;
static constexpr int NTOK  = Bn * KEEP;               // 1728 tokens
static constexpr int XZN   = 2 * DI;                  // 1536
static constexpr int XDBC  = 64;                      // padded 56 -> 64
static constexpr int DTK   = 32;                      // padded 24 -> 32

// ---------------- WMMA vector types ----------------
typedef __attribute__((ext_vector_type(16))) __bf16 v16bf;
typedef __attribute__((ext_vector_type(8)))  __bf16 v8bf;
typedef __attribute__((ext_vector_type(8)))  float  v8f;

__device__ __forceinline__ __bf16 f2bf(float f) {
    union { float f; unsigned u; } in; in.f = f;
    unsigned u = in.u;
    unsigned r = (u + 0x7FFFu + ((u >> 16) & 1u)) >> 16;   // round-nearest-even
    union { unsigned short s; __bf16 b; } out; out.s = (unsigned short)r;
    return out.b;
}

__device__ __forceinline__ float siluf(float x) { return x / (1.f + expf(-x)); }

// ---------------- mask prefix sum (replaces stable argsort) ----------------
__global__ void k_rank(const int* __restrict__ active,
                       int* __restrict__ rank, int* __restrict__ idsk)
{
    int b = blockIdx.x;
    if (threadIdx.x != 0) return;
    int cnt = 0;
    for (int l = 0; l < Lt; ++l) {
        if (active[b * Lt + l] != 0) {
            if (cnt < KEEP) idsk[b * KEEP + cnt] = l;
            rank[b * Lt + l] = cnt;
            ++cnt;
        } else {
            rank[b * Lt + l] = -1;
        }
    }
}

// ---------------- gather kept tokens + pos embed ----------------
__global__ void k_gather(const float* __restrict__ imgs, const float* __restrict__ pos,
                         const int* __restrict__ idsk, float* __restrict__ x)
{
    int idx = blockIdx.x * blockDim.x + threadIdx.x;
    if (idx >= NTOK * DM) return;
    int d = idx % DM, t = idx / DM;
    int b = t / KEEP;
    int l = idsk[t];
    x[idx] = imgs[((size_t)b * DM + d) * Lt + l] + pos[(size_t)l * DM + d];
}

// ---------------- generic f32 -> bf16 convert with zero-padding ----------------
__global__ void k_cvt(const float* __restrict__ src, __bf16* __restrict__ dst,
                      int dstR, int dstC, int srcStride, int copyR, int copyC)
{
    int idx = blockIdx.x * blockDim.x + threadIdx.x;
    if (idx >= dstR * dstC) return;
    int r = idx / dstC, c = idx % dstC;
    float v = (r < copyR && c < copyC) ? src[(size_t)r * srcStride + c] : 0.f;
    dst[idx] = f2bf(v);
}

// ---------------- LayerNorm (one wave32 per token row), writes bf16 ----------------
__global__ void k_ln(const float* __restrict__ x, const float* __restrict__ w,
                     const float* __restrict__ b, __bf16* __restrict__ out, int cols)
{
    int row  = blockIdx.x * (blockDim.x >> 5) + (threadIdx.x >> 5);
    int lane = threadIdx.x & 31;
    const float* xr = x + (size_t)row * cols;
    float s = 0.f, s2 = 0.f;
    for (int c = lane; c < cols; c += 32) { float v = xr[c]; s += v; s2 += v * v; }
    for (int m = 16; m; m >>= 1) { s += __shfl_xor(s, m, 32); s2 += __shfl_xor(s2, m, 32); }
    float mu  = s / cols;
    float var = s2 / cols - mu * mu;
    float inv = rsqrtf(var + 1e-5f);
    for (int c = lane; c < cols; c += 32)
        out[(size_t)row * cols + c] = f2bf((xr[c] - mu) * inv * w[c] + b[c]);
}

// ---------------- WMMA GEMM: C[M,N] = A[M,K](bf16) * W[N,K]^T(bf16) (+bias)(+addend) ----
// K is a compile-time constant: all W-row / K-step addressing folds into the
// 24-bit VGLOBAL immediate offset (2 live address pairs, no spill), and the
// ping-pong double buffer is unrolled so there are no register copies: even
// K-steps consume buffer 0 while buffer 1's global_load_b128 clause is in
// flight, and vice versa.
template<int K>
__global__ void k_gemm(const __bf16* __restrict__ A, const __bf16* __restrict__ W,
                       const float* __restrict__ bias, const float* __restrict__ addend,
                       float* __restrict__ C, int Mr, int Nc)
{
    constexpr int STEPS = K / 32;
    int wid  = (blockIdx.x * blockDim.x + threadIdx.x) >> 5;
    int lane = threadIdx.x & 31;
    int tilesN = Nc >> 6;
    int totalW = (Mr >> 4) * tilesN;
    if (wid >= totalW) return;
    int mt = wid / tilesN, nt = wid % tilesN;
    int half = lane >> 4, ml = lane & 15;

    // Fragment addressing used here (bf16 16x16x32, contiguous k order, same for A and B):
    //   A 16x32: lane m holds cols kk + 16*half .. +15 (one contiguous run)
    //   B 32x16: lane n holds cols kk + 16*half .. +15 (one contiguous run)
    const __bf16* Ap = A + (size_t)(mt * 16 + ml) * K + 16 * half;
    const __bf16* Wp = W + (size_t)(nt * 64 + ml) * K + 16 * half;

    union Afrag { v16bf v; v8bf h[2]; };
    Afrag a[2];
    v16bf b[2][4];
    v8f acc[4] = {v8f{}, v8f{}, v8f{}, v8f{}};

    auto loadAB = [&](int buf, int kk) {
        a[buf].h[0] = *(const v8bf*)(Ap + kk);
        a[buf].h[1] = *(const v8bf*)(Ap + kk + 8);
#pragma unroll
        for (int j = 0; j < 4; ++j)
            b[buf][j] = *(const v16bf*)(Wp + j * 16 * K + kk);   // immediate offsets
    };
    auto mma = [&](int buf) {
#pragma unroll
        for (int j = 0; j < 4; ++j)
            acc[j] = __builtin_amdgcn_wmma_f32_16x16x32_bf16(
                         false, a[buf].v, false, b[buf][j], (short)0, acc[j], false, false);
    };

    loadAB(0, 0);
#pragma unroll
    for (int s = 0; s < STEPS; s += 2) {
        if (s + 1 < STEPS) {
            loadAB(1, (s + 1) * 32);                             // in flight behind mma(0)
            __builtin_prefetch(Ap + (s + 1) * 32, 0, 3);         // global_prefetch_b8
        }
        mma(0);
        if (s + 1 < STEPS) {
            if (s + 2 < STEPS) loadAB(0, (s + 2) * 32);          // in flight behind mma(1)
            mma(1);
        }
    }

#pragma unroll
    for (int j = 0; j < 4; ++j) {
        int col = nt * 64 + j * 16 + ml;
        float bv = bias ? bias[col] : 0.f;
#pragma unroll
        for (int v = 0; v < 8; ++v) {   // C layout: row = 16*mt + v + 8*half, col = lane&15
            int row = mt * 16 + v + 8 * half;
            float o = acc[j][v] + bv;
            if (addend) o += addend[(size_t)row * Nc + col];
            C[(size_t)row * Nc + col] = o;
        }
    }
}

// ---------------- causal depthwise conv (per scan direction) + SiLU ----------------
__global__ void k_conv(const float* __restrict__ xz, const float* __restrict__ cw,
                       const float* __restrict__ cb, int dir,
                       float* __restrict__ xc, __bf16* __restrict__ xcb)
{
    int idx = blockIdx.x * blockDim.x + threadIdx.x;
    if (idx >= NTOK * DI) return;
    int d = idx % DI, r = idx / DI;
    int b = r / KEEP, p = r % KEEP;
    float acc = 0.f;
#pragma unroll
    for (int k = 0; k < DC; ++k) {
        int q = p + k - (DC - 1);
        if (q >= 0) {
            int o = dir ? (KEEP - 1 - q) : q;                 // scan-order -> original index
            acc += cw[d * DC + k] * xz[(size_t)(b * KEEP + o) * XZN + d];
        }
    }
    float s = siluf(acc + cb[d]);
    xc[idx]  = s;
    xcb[idx] = f2bf(s);
}

// ---------------- selective scan: thread = (batch, channel); 16 states in VGPRs ------
__global__ void k_scan(const float* __restrict__ dp,   // (NTOK, DI) delta pre-activation
                       const float* __restrict__ xc,   // (NTOK, DI) conv output
                       const float* __restrict__ xdb,  // (NTOK, 64): [24..39]=B, [40..55]=C
                       const float* __restrict__ dtb,  // (DI)
                       const float* __restrict__ a_log,// (DI, NS)
                       float* __restrict__ ys)         // (NTOK, DI)
{
    __shared__ float sBC[2 * NS];
    int b = blockIdx.x / (DI / 256);
    int chunk = blockIdx.x % (DI / 256);
    int d = chunk * 256 + threadIdx.x;

    float Ar[NS], h[NS];
#pragma unroll
    for (int n = 0; n < NS; ++n) { Ar[n] = -expf(a_log[(size_t)d * NS + n]); h[n] = 0.f; }
    float bias = dtb[d];

    for (int p = 0; p < KEEP; ++p) {
        int r = b * KEEP + p;
        if (threadIdx.x < 2 * NS) sBC[threadIdx.x] = xdb[(size_t)r * XDBC + 24 + threadIdx.x];
        __syncthreads();
        float xv = dp[(size_t)r * DI + d] + bias;
        float delta = (xv > 20.f) ? xv : log1pf(expf(xv));     // softplus
        float u = xc[(size_t)r * DI + d];
        float y = 0.f;
#pragma unroll
        for (int n = 0; n < NS; ++n) {
            float dA = expf(delta * Ar[n]);
            h[n] = dA * h[n] + delta * sBC[n] * u;
            y += h[n] * sBC[NS + n];
        }
        ys[(size_t)r * DI + d] = y;
        __syncthreads();
    }
}

// ---------------- combine both directions, D-skip, gate by silu(z) ----------------
__global__ void k_combine(const float* __restrict__ ysf, const float* __restrict__ ysb,
                          const float* __restrict__ xcf, const float* __restrict__ xcb,
                          const float* __restrict__ xz,  const float* __restrict__ dskip,
                          __bf16* __restrict__ ybf)
{
    int idx = blockIdx.x * blockDim.x + threadIdx.x;
    if (idx >= NTOK * DI) return;
    int d = idx % DI, t = idx / DI;
    int b = t / KEEP, po = t % KEEP;
    int rb = b * KEEP + (KEEP - 1 - po);                       // backward scan row
    float z  = xz[(size_t)t * XZN + DI + d];
    float sz = siluf(z);
    float ds = dskip[d];
    float v = (ysf[(size_t)t * DI + d] + ds * xcf[(size_t)t * DI + d] +
               ysb[(size_t)rb * DI + d] + ds * xcb[(size_t)rb * DI + d]) * sz;
    ybf[idx] = f2bf(v);
}

// ---------------- exact GELU -> bf16 ----------------
__global__ void k_gelu(const float* __restrict__ in, __bf16* __restrict__ out, int n)
{
    int idx = blockIdx.x * blockDim.x + threadIdx.x;
    if (idx >= n) return;
    float x = in[idx];
    out[idx] = f2bf(0.5f * x * (1.f + erff(x * 0.70710678118654752f)));
}

// ---------------- scatter back: kept tokens + mask tokens, to (B, DM, L) ----------------
__global__ void k_scatter(const float* __restrict__ x, const int* __restrict__ rank,
                          const float* __restrict__ mask_token, float* __restrict__ out)
{
    int idx = blockIdx.x * blockDim.x + threadIdx.x;
    if (idx >= Bn * DM * Lt) return;
    int l = idx % Lt, d = (idx / Lt) % DM, b = idx / (Lt * DM);
    int rk = rank[b * Lt + l];
    out[idx] = (rk >= 0) ? x[((size_t)(b * KEEP + rk)) * DM + d] : mask_token[d];
}

// =====================================================================
static inline int gblk(long long n, int t = 256) { return (int)((n + t - 1) / t); }

extern "C" void kernel_launch(void* const* d_in, const int* in_sizes, int n_in,
                              void* d_out, int out_size, void* d_ws, size_t ws_size,
                              hipStream_t stream)
{
    const float* imgs      = (const float*)d_in[0];
    const float* pos       = (const float*)d_in[1];
    const float* mask_tok  = (const float*)d_in[2];
    const float* n1w       = (const float*)d_in[3];
    const float* n1b       = (const float*)d_in[4];
    const float* inW       = (const float*)d_in[5];
    const float* convW     = (const float*)d_in[6];
    const float* convBi    = (const float*)d_in[7];
    const float* xpW       = (const float*)d_in[8];
    const float* dtW       = (const float*)d_in[9];
    const float* dtB       = (const float*)d_in[10];
    const float* Alog      = (const float*)d_in[11];
    const float* Ablog     = (const float*)d_in[12];
    const float* Dskip     = (const float*)d_in[13];
    const float* outW      = (const float*)d_in[14];
    const float* n2w       = (const float*)d_in[15];
    const float* n2b       = (const float*)d_in[16];
    const float* f1W       = (const float*)d_in[17];
    const float* f1B       = (const float*)d_in[18];
    const float* f2W       = (const float*)d_in[19];
    const float* f2B       = (const float*)d_in[20];
    const int*   active    = (const int*)d_in[21];
    float* out = (float*)d_out;

    // ---- workspace bump allocator (256B aligned) ----
    char* base = (char*)d_ws;
    size_t off = 0;
    auto alloc = [&](size_t bytes) -> void* {
        off = (off + 255) & ~(size_t)255;
        void* p = base + off; off += bytes; return p;
    };
    int*    rank   = (int*)   alloc((size_t)Bn * Lt * 4);
    int*    idsk   = (int*)   alloc((size_t)Bn * KEEP * 4);
    float*  x      = (float*) alloc((size_t)NTOK * DM * 4);
    __bf16* h_bf   = (__bf16*)alloc((size_t)NTOK * DM * 2);
    float*  xz     = (float*) alloc((size_t)NTOK * XZN * 4);
    float*  xcf    = (float*) alloc((size_t)NTOK * DI * 4);
    float*  xcb    = (float*) alloc((size_t)NTOK * DI * 4);
    __bf16* xcf_bf = (__bf16*)alloc((size_t)NTOK * DI * 2);
    __bf16* xcb_bf = (__bf16*)alloc((size_t)NTOK * DI * 2);
    float*  xdb    = (float*) alloc((size_t)NTOK * XDBC * 4);
    __bf16* dtr_bf = (__bf16*)alloc((size_t)NTOK * DTK * 2);
    float*  dp     = (float*) alloc((size_t)NTOK * DI * 4);
    float*  ysf    = (float*) alloc((size_t)NTOK * DI * 4);
    float*  ysb    = (float*) alloc((size_t)NTOK * DI * 4);
    __bf16* y_bf   = (__bf16*)alloc((size_t)NTOK * DI * 2);
    float*  mlp    = (float*) alloc((size_t)NTOK * 2 * DM * 4);
    __bf16* mlp_bf = (__bf16*)alloc((size_t)NTOK * 2 * DM * 2);
    // bf16 weight staging (reused each layer)
    __bf16* win    = (__bf16*)alloc((size_t)XZN * DM * 2);
    __bf16* wxp    = (__bf16*)alloc((size_t)XDBC * DI * 2);
    __bf16* wdt    = (__bf16*)alloc((size_t)DI * DTK * 2);
    __bf16* wout   = (__bf16*)alloc((size_t)DM * DI * 2);
    __bf16* wf1    = (__bf16*)alloc((size_t)2 * DM * DM * 2);
    __bf16* wf2    = (__bf16*)alloc((size_t)DM * 2 * DM * 2);
    (void)ws_size; (void)in_sizes; (void)n_in; (void)out_size;

    auto gemm = [&](const __bf16* A, const __bf16* W, const float* bias,
                    const float* addend, float* C, int Mr, int Nc, int K) {
        long long waves = (long long)(Mr / 16) * (Nc / 64);
        dim3 g(gblk(waves * 32)), blk(256);
        if (K == 384)
            hipLaunchKernelGGL(k_gemm<384>, g, blk, 0, stream, A, W, bias, addend, C, Mr, Nc);
        else if (K == 768)
            hipLaunchKernelGGL(k_gemm<768>, g, blk, 0, stream, A, W, bias, addend, C, Mr, Nc);
        else
            hipLaunchKernelGGL(k_gemm<32>,  g, blk, 0, stream, A, W, bias, addend, C, Mr, Nc);
    };
    auto cvt = [&](const float* src, __bf16* dst, int dR, int dC, int sS, int cR, int cC) {
        hipLaunchKernelGGL(k_cvt, dim3(gblk((long long)dR * dC)), dim3(256), 0, stream,
                           src, dst, dR, dC, sS, cR, cC);
    };

    // ---- token selection + gather ----
    hipLaunchKernelGGL(k_rank, dim3(Bn), dim3(32), 0, stream, active, rank, idsk);
    hipLaunchKernelGGL(k_gather, dim3(gblk((long long)NTOK * DM)), dim3(256), 0, stream,
                       imgs, pos, idsk, x);

    for (int l = 0; l < DEPTH; ++l) {
        // weight staging to bf16 (with K/N padding where needed)
        cvt(inW  + (size_t)l * XZN * DM, win,  XZN,  DM,  DM,  XZN, DM);
        cvt(xpW  + (size_t)l * (DTR + 2 * NS) * DI, wxp, XDBC, DI, DI, DTR + 2 * NS, DI);
        cvt(dtW  + (size_t)l * DI * DTR, wdt, DI, DTK, DTR, DI, DTR);
        cvt(outW + (size_t)l * DM * DI,  wout, DM,  DI,  DI,  DM,  DI);
        cvt(f1W  + (size_t)l * 2 * DM * DM, wf1, 2 * DM, DM, DM, 2 * DM, DM);
        cvt(f2W  + (size_t)l * DM * 2 * DM, wf2, DM, 2 * DM, 2 * DM, DM, 2 * DM);

        // LN1 -> in_proj
        hipLaunchKernelGGL(k_ln, dim3(NTOK / 8), dim3(256), 0, stream,
                           x, n1w + (size_t)l * DM, n1b + (size_t)l * DM, h_bf, DM);
        gemm(h_bf, win, nullptr, nullptr, xz, NTOK, XZN, DM);

        // depthwise conv, both directions
        hipLaunchKernelGGL(k_conv, dim3(gblk((long long)NTOK * DI)), dim3(256), 0, stream,
                           xz, convW + (size_t)l * DI * DC, convBi + (size_t)l * DI, 0, xcf, xcf_bf);
        hipLaunchKernelGGL(k_conv, dim3(gblk((long long)NTOK * DI)), dim3(256), 0, stream,
                           xz, convW + (size_t)l * DI * DC, convBi + (size_t)l * DI, 1, xcb, xcb_bf);

        // per-direction: x_proj -> dt_proj -> selective scan
        for (int dir = 0; dir < 2; ++dir) {
            const __bf16* xc_b = dir ? xcb_bf : xcf_bf;
            const float*  xc_f = dir ? xcb : xcf;
            float* ys          = dir ? ysb : ysf;
            const float* alog  = (dir ? Ablog : Alog) + (size_t)l * DI * NS;
            gemm(xc_b, wxp, nullptr, nullptr, xdb, NTOK, XDBC, DI);
            cvt(xdb, dtr_bf, NTOK, DTK, XDBC, NTOK, DTR);          // extract + pad dtr
            gemm(dtr_bf, wdt, nullptr, nullptr, dp, NTOK, DI, DTK);
            hipLaunchKernelGGL(k_scan, dim3(Bn * (DI / 256)), dim3(256), 0, stream,
                               dp, xc_f, xdb, dtB + (size_t)l * DI, alog, ys);
        }

        // combine + gate, out_proj with residual
        hipLaunchKernelGGL(k_combine, dim3(gblk((long long)NTOK * DI)), dim3(256), 0, stream,
                           ysf, ysb, xcf, xcb, xz, Dskip + (size_t)l * DI, y_bf);
        gemm(y_bf, wout, nullptr, x, x, NTOK, DM, DI);

        // LN2 -> MLP with residual
        hipLaunchKernelGGL(k_ln, dim3(NTOK / 8), dim3(256), 0, stream,
                           x, n2w + (size_t)l * DM, n2b + (size_t)l * DM, h_bf, DM);
        gemm(h_bf, wf1, f1B + (size_t)l * 2 * DM, nullptr, mlp, NTOK, 2 * DM, DM);
        hipLaunchKernelGGL(k_gelu, dim3(gblk((long long)NTOK * 2 * DM)), dim3(256), 0, stream,
                           mlp, mlp_bf, NTOK * 2 * DM);
        gemm(mlp_bf, wf2, f2B + (size_t)l * DM, x, x, NTOK, DM, 2 * DM);
    }

    // ---- scatter back to (B, DM, 12, 12, 12) ----
    hipLaunchKernelGGL(k_scatter, dim3(gblk((long long)Bn * DM * Lt)), dim3(256), 0, stream,
                       x, rank, mask_tok, out);
}
